// GNNRecommender_41334765256797
// MI455X (gfx1250) — matrix operand order repaired and checked by reading, not compiled, not run on hardware
//
#include <hip/hip_runtime.h>
#include <hip/hip_bf16.h>

typedef __attribute__((ext_vector_type(16))) _Float16 v16h;
typedef __attribute__((ext_vector_type(8)))  _Float16 v8h;
typedef __attribute__((ext_vector_type(8)))  float    v8f;

#define HD   128
#define EMBD 64
#define TT   20

// ---------------------------------------------------------------------------
// WMMA helpers (CDNA5 wave32, V_WMMA_F32_16X16X32_F16 layouts per ISA 7.12.2)
// ---------------------------------------------------------------------------
__device__ __forceinline__ v8f wmma_f16(v16h a, v16h b, v8f c) {
  // (neg_a, A, neg_b, B, c_mod, C, reuse_a, reuse_b)
  return __builtin_amdgcn_wmma_f32_16x16x32_f16(false, a, false, b, (short)0, c,
                                                false, false);
}

// A fragment (16x32 MxK) from a row-major f32 matrix, with per-row scale.
// lane l: row = l&15 ; halfs j<8 -> K = kbase + (l>>4)*8 + j
//                      halfs j>=8 -> K = kbase + (l>>4)*8 + 16 + (j-8)
__device__ __forceinline__ v16h loadA_f32(const float* M, int ldm, int rowbase,
                                          int kbase, int lane, float scale) {
  const float* p = M + (size_t)(rowbase + (lane & 15)) * ldm + kbase +
                   ((lane >> 4) << 3);
  v16h a;
#pragma unroll
  for (int j = 0; j < 8; ++j) {
    a[j]     = (_Float16)(p[j] * scale);
    a[j + 8] = (_Float16)(p[j + 16] * scale);
  }
  return a;
}

// A fragment from a row-major f16 matrix in global memory (two 16B loads).
__device__ __forceinline__ v16h loadA_g16(const _Float16* M, int ldm,
                                          int rowbase, int kbase, int lane) {
  const _Float16* p = M + (size_t)(rowbase + (lane & 15)) * ldm + kbase +
                      ((lane >> 4) << 3);
  v8h lo = *(const v8h*)p;
  v8h hi = *(const v8h*)(p + 16);
  v16h a;
#pragma unroll
  for (int j = 0; j < 8; ++j) { a[j] = lo[j]; a[j + 8] = hi[j]; }
  return a;
}

// B fragment (32x16 KxN) of W.T where W is row-major f16 [n][k] (out = A@W.T).
// lane l: N = nbase + (l&15) ; halfs j<8 -> K = kbase + (l>>4)*8 + j
//                              halfs j>=8 -> K = kbase + (l>>4)*8 + 16 + (j-8)
__device__ __forceinline__ v16h loadB_w16(const _Float16* W, int ldk, int nbase,
                                          int kbase, int lane) {
  const _Float16* p = W + (size_t)(nbase + (lane & 15)) * ldk + kbase +
                      ((lane >> 4) << 3);
  v8h lo = *(const v8h*)p;
  v8h hi = *(const v8h*)(p + 16);
  v16h b;
#pragma unroll
  for (int j = 0; j < 8; ++j) { b[j] = lo[j]; b[j + 8] = hi[j]; }
  return b;
}

// A fragment from an LDS f16 buffer laid out [16 rows][HD halfs].
__device__ __forceinline__ v16h loadA_lds16(const _Float16* hb, int kbase,
                                            int lane) {
  const _Float16* p = hb + (lane & 15) * HD + kbase + ((lane >> 4) << 3);
  v8h lo = *(const v8h*)p;
  v8h hi = *(const v8h*)(p + 16);
  v16h a;
#pragma unroll
  for (int j = 0; j < 8; ++j) { a[j] = lo[j]; a[j + 8] = hi[j]; }
  return a;
}

__device__ __forceinline__ float sigmoidf_dev(float x) {
  return 1.0f / (1.0f + __expf(-x));
}

// ---------------------------------------------------------------------------
// f32 -> f16 conversion (weights + optional x_user pre-convert)
// ---------------------------------------------------------------------------
__global__ void cvt_f16(const float* __restrict__ s, _Float16* __restrict__ d,
                        int n) {
  int i = blockIdx.x * blockDim.x + threadIdx.x;
  if (i < n) d[i] = (_Float16)s[i];
}

// ---------------------------------------------------------------------------
// Edge degree counting + feature scatter-add (mean aggregation numerator)
// ---------------------------------------------------------------------------
__global__ void edge_count(const int* __restrict__ idx, float* __restrict__ cnt,
                           long long n) {
  long long i = (long long)blockIdx.x * blockDim.x + threadIdx.x;
  if (i < n) atomicAdd(&cnt[idx[i]], 1.0f);
}

__global__ void edge_scatter(const int* __restrict__ sidx,
                             const int* __restrict__ didx,
                             const float* __restrict__ srcX,
                             float* __restrict__ agg, long long nE) {
  long long g = (long long)blockIdx.x * blockDim.x + threadIdx.x;
  long long e = g >> 5;              // 32 threads per edge, 4 channels each
  if (e >= nE) return;
  int ch = (int)(g & 31) * 4;
  int s = sidx[e], d = didx[e];
  const float4 v = *(const float4*)(srcX + (size_t)s * HD + ch);
  float* o = agg + (size_t)d * HD + ch;
  atomicAdd(o + 0, v.x);
  atomicAdd(o + 1, v.y);
  atomicAdd(o + 2, v.z);
  atomicAdd(o + 3, v.w);
}

// ---------------------------------------------------------------------------
// Fused LSTM: 1 block = 16 users, 4 waves, wave w owns gate w (i,f,g,o).
// Per step: gates = x_t @ Wih.T + h @ Whh.T (WMMA), exchange via LDS,
// elementwise c/h update, h kept in LDS as f16. Weight B-fragments are
// loop-invariant and get hoisted to (high) VGPRs by the compiler.
// ---------------------------------------------------------------------------
template <bool X16>
__global__ void __launch_bounds__(128)
lstm_kernel(const float* __restrict__ x, const _Float16* __restrict__ x16,
            const _Float16* __restrict__ Wih, const _Float16* __restrict__ Whh,
            const float* __restrict__ bih, const float* __restrict__ bhh,
            float* __restrict__ uout) {
  __shared__ _Float16 hbuf[16 * HD];     // 4 KB
  __shared__ float    cbuf[16 * HD];     // 8 KB
  __shared__ float    gbuf[4 * 16 * HD]; // 32 KB
  const int tid = threadIdx.x, lane = tid & 31, wave = tid >> 5;
  const int rowbase = blockIdx.x * 16;
  for (int i = tid; i < 16 * HD; i += 128) {
    hbuf[i] = (_Float16)0.0f;
    cbuf[i] = 0.0f;
  }
  __syncthreads();
  const int nlo = lane & 15, mhi = (lane >> 4) << 3;
  for (int t = 0; t < TT; ++t) {
    v8f acc[8] = {};
    // x_t @ Wih.T  (K = EMB = 64)
#pragma unroll
    for (int ks = 0; ks < 2; ++ks) {
      v16h a;
      if (X16)
        a = loadA_g16(x16 + t * EMBD, TT * EMBD, rowbase, ks * 32, lane);
      else
        a = loadA_f32(x + t * EMBD, TT * EMBD, rowbase, ks * 32, lane, 1.0f);
#pragma unroll
      for (int nt = 0; nt < 8; ++nt) {
        v16h b = loadB_w16(Wih, EMBD, wave * HD + nt * 16, ks * 32, lane);
        acc[nt] = wmma_f16(a, b, acc[nt]);
      }
    }
    // h @ Whh.T  (K = H = 128)
#pragma unroll
    for (int ks = 0; ks < 4; ++ks) {
      v16h a = loadA_lds16(hbuf, ks * 32, lane);
#pragma unroll
      for (int nt = 0; nt < 8; ++nt) {
        v16h b = loadB_w16(Whh, HD, wave * HD + nt * 16, ks * 32, lane);
        acc[nt] = wmma_f16(a, b, acc[nt]);
      }
    }
    // Write this wave's gate tile (16x128 f32) to LDS.
    float* g = gbuf + wave * 16 * HD;
#pragma unroll
    for (int nt = 0; nt < 8; ++nt)
#pragma unroll
      for (int r = 0; r < 8; ++r)
        g[(mhi + r) * HD + nt * 16 + nlo] = acc[nt][r];
    __syncthreads();
    // Elementwise gate math: 128 threads x 16 (user,col) pairs.
    for (int idx = tid; idx < 16 * HD; idx += 128) {
      int n = idx & (HD - 1);
      float iv = gbuf[0 * 16 * HD + idx] + bih[n] + bhh[n];
      float fv = gbuf[1 * 16 * HD + idx] + bih[HD + n] + bhh[HD + n];
      float gv = gbuf[2 * 16 * HD + idx] + bih[2 * HD + n] + bhh[2 * HD + n];
      float ov = gbuf[3 * 16 * HD + idx] + bih[3 * HD + n] + bhh[3 * HD + n];
      float cc = sigmoidf_dev(fv) * cbuf[idx] + sigmoidf_dev(iv) * tanhf(gv);
      float hh = sigmoidf_dev(ov) * tanhf(cc);
      cbuf[idx] = cc;
      hbuf[idx] = (_Float16)hh;
      if (t == TT - 1) {
        int m = idx >> 7;
        uout[(size_t)(rowbase + m) * HD + n] = hh;
      }
    }
    __syncthreads();
  }
}

// ---------------------------------------------------------------------------
// out[N x 128] = A[N x K] @ W.T + bias  (optional ReLU). 1 wave = 16 rows.
// ---------------------------------------------------------------------------
__global__ void __launch_bounds__(128)
linear128(const float* __restrict__ A, int K, const _Float16* __restrict__ W,
          const float* __restrict__ bias, float* __restrict__ out, int doRelu) {
  const int lane = threadIdx.x & 31, wave = threadIdx.x >> 5;
  const int rowbase = (blockIdx.x * 4 + wave) * 16;
  v8f acc[8] = {};
  for (int ks = 0; ks < K / 32; ++ks) {
    v16h a = loadA_f32(A, K, rowbase, ks * 32, lane, 1.0f);
#pragma unroll
    for (int nt = 0; nt < 8; ++nt)
      acc[nt] = wmma_f16(a, loadB_w16(W, K, nt * 16, ks * 32, lane), acc[nt]);
  }
  const int nlo = lane & 15, mhi = (lane >> 4) << 3;
#pragma unroll
  for (int nt = 0; nt < 8; ++nt)
#pragma unroll
    for (int r = 0; r < 8; ++r) {
      int n = nt * 16 + nlo;
      float v = acc[nt][r] + bias[n];
      if (doRelu) v = fmaxf(v, 0.0f);
      out[(size_t)(rowbase + mhi + r) * HD + n] = v;
    }
}

// ---------------------------------------------------------------------------
// SAGE transform: out = relu((agg/max(cnt,1)) @ Wl.T + bl + dstX @ Wr.T)
// Mean-divide fused into the A-fragment convert (per-row scale).
// ---------------------------------------------------------------------------
__global__ void __launch_bounds__(128)
sage_gemm(const float* __restrict__ agg, const float* __restrict__ cnt,
          const float* __restrict__ dstX, const _Float16* __restrict__ Wl,
          const _Float16* __restrict__ Wr, const float* __restrict__ bl,
          float* __restrict__ out) {
  const int lane = threadIdx.x & 31, wave = threadIdx.x >> 5;
  const int rowbase = (blockIdx.x * 4 + wave) * 16;
  const float inv = 1.0f / fmaxf(cnt[rowbase + (lane & 15)], 1.0f);
  v8f acc[8] = {};
#pragma unroll
  for (int ks = 0; ks < 4; ++ks) {
    v16h a = loadA_f32(agg, HD, rowbase, ks * 32, lane, inv);
#pragma unroll
    for (int nt = 0; nt < 8; ++nt)
      acc[nt] = wmma_f16(a, loadB_w16(Wl, HD, nt * 16, ks * 32, lane), acc[nt]);
  }
#pragma unroll
  for (int ks = 0; ks < 4; ++ks) {
    v16h a = loadA_f32(dstX, HD, rowbase, ks * 32, lane, 1.0f);
#pragma unroll
    for (int nt = 0; nt < 8; ++nt)
      acc[nt] = wmma_f16(a, loadB_w16(Wr, HD, nt * 16, ks * 32, lane), acc[nt]);
  }
  const int nlo = lane & 15, mhi = (lane >> 4) << 3;
#pragma unroll
  for (int nt = 0; nt < 8; ++nt)
#pragma unroll
    for (int r = 0; r < 8; ++r) {
      int n = nt * 16 + nlo;
      out[(size_t)(rowbase + mhi + r) * HD + n] =
          fmaxf(acc[nt][r] + bl[n], 0.0f);
    }
}

// ---------------------------------------------------------------------------
// Classifier: h1 = relu([u,c] @ W1.T + b1) ; out = sigmoid(h1 . w2 + b2)
// ---------------------------------------------------------------------------
__global__ void __launch_bounds__(128)
classifier(const float* __restrict__ u, const float* __restrict__ c,
           const _Float16* __restrict__ W1, const float* __restrict__ b1,
           const float* __restrict__ w2, const float* __restrict__ b2,
           float* __restrict__ out) {
  __shared__ float h1s[4 * 16 * HD]; // 32 KB
  const int lane = threadIdx.x & 31, wave = threadIdx.x >> 5;
  const int rowbase = (blockIdx.x * 4 + wave) * 16;
  v8f acc[8] = {};
#pragma unroll
  for (int ks = 0; ks < 4; ++ks) {      // K 0..127 from u
    v16h a = loadA_f32(u, HD, rowbase, ks * 32, lane, 1.0f);
#pragma unroll
    for (int nt = 0; nt < 8; ++nt)
      acc[nt] =
          wmma_f16(a, loadB_w16(W1, 2 * HD, nt * 16, ks * 32, lane), acc[nt]);
  }
#pragma unroll
  for (int ks = 0; ks < 4; ++ks) {      // K 128..255 from c
    v16h a = loadA_f32(c, HD, rowbase, ks * 32, lane, 1.0f);
#pragma unroll
    for (int nt = 0; nt < 8; ++nt)
      acc[nt] = wmma_f16(a, loadB_w16(W1, 2 * HD, nt * 16, HD + ks * 32, lane),
                         acc[nt]);
  }
  const int nlo = lane & 15, mhi = (lane >> 4) << 3;
  float* hw = h1s + wave * 16 * HD;
#pragma unroll
  for (int nt = 0; nt < 8; ++nt)
#pragma unroll
    for (int r = 0; r < 8; ++r) {
      int n = nt * 16 + nlo;
      hw[(mhi + r) * HD + n] = fmaxf(acc[nt][r] + b1[n], 0.0f);
    }
  __syncthreads();
  // Final dot with w2: lane pair (l, l+16) splits the 128 columns of row l&15.
  const int row = lane & 15, nb = (lane >> 4) * 64;
  float s = 0.0f;
  for (int n = 0; n < 64; ++n) s += hw[row * HD + nb + n] * w2[nb + n];
  s += __shfl_xor(s, 16, 32);
  if (lane < 16)
    out[rowbase + row] = sigmoidf_dev(s + b2[0]);
}

// ---------------------------------------------------------------------------
// Host orchestration
// ---------------------------------------------------------------------------
extern "C" void kernel_launch(void* const* d_in, const int* in_sizes, int n_in,
                              void* d_out, int out_size, void* d_ws,
                              size_t ws_size, hipStream_t stream) {
  const float* x_user    = (const float*)d_in[0];
  const float* x_content = (const float*)d_in[1];
  const int*   edges     = (const int*)d_in[2];
  const float* W_ih = (const float*)d_in[3];
  const float* W_hh = (const float*)d_in[4];
  const float* b_ih = (const float*)d_in[5];
  const float* b_hh = (const float*)d_in[6];
  const float* Wc   = (const float*)d_in[7];
  const float* bc   = (const float*)d_in[8];
  const float* Wl0  = (const float*)d_in[9];
  const float* bl0  = (const float*)d_in[10];
  const float* Wr0  = (const float*)d_in[11];
  const float* Wl1  = (const float*)d_in[12];
  const float* bl1  = (const float*)d_in[13];
  const float* Wr1  = (const float*)d_in[14];
  const float* W1   = (const float*)d_in[15];
  const float* b1   = (const float*)d_in[16];
  const float* W2   = (const float*)d_in[17];
  const float* b2   = (const float*)d_in[18];
  (void)n_in; (void)out_size;

  const int N = in_sizes[1] / EMBD;           // 32768
  const long long E = in_sizes[2] / 2;        // 1,048,576
  const int* srcI = edges;
  const int* dstI = edges + E;

  char* w = (char*)d_ws;
  auto take = [&](size_t bytes) {
    char* p = w;
    w += (bytes + 255) & ~(size_t)255;
    return p;
  };
  float* u    = (float*)take((size_t)N * HD * 4);
  float* c    = (float*)take((size_t)N * HD * 4);
  float* u2   = (float*)take((size_t)N * HD * 4);
  float* c2   = (float*)take((size_t)N * HD * 4);
  float* agg  = (float*)take((size_t)N * HD * 4);
  float* cntS = (float*)take((size_t)N * 4);
  float* cntD = (float*)take((size_t)N * 4);
  _Float16* wih = (_Float16*)take((size_t)4 * HD * EMBD * 2);
  _Float16* whh = (_Float16*)take((size_t)4 * HD * HD * 2);
  _Float16* wc  = (_Float16*)take((size_t)HD * EMBD * 2);
  _Float16* wl0 = (_Float16*)take((size_t)HD * HD * 2);
  _Float16* wr0 = (_Float16*)take((size_t)HD * HD * 2);
  _Float16* wl1 = (_Float16*)take((size_t)HD * HD * 2);
  _Float16* wr1 = (_Float16*)take((size_t)HD * HD * 2);
  _Float16* w1h = (_Float16*)take((size_t)HD * 2 * HD * 2);

  // Optional: pre-convert x_user to f16 (halves LSTM x traffic, removes cvts
  // from the recurrent critical path). Deterministic: depends only on ws_size.
  const size_t xbytes = (size_t)N * TT * EMBD * 2;
  _Float16* x16 = nullptr;
  if ((size_t)(w - (char*)d_ws) + xbytes + 256 <= ws_size)
    x16 = (_Float16*)take(xbytes);

  auto cvt = [&](const float* s, _Float16* d, int n) {
    cvt_f16<<<(n + 255) / 256, 256, 0, stream>>>(s, d, n);
  };
  cvt(W_ih, wih, 4 * HD * EMBD);
  cvt(W_hh, whh, 4 * HD * HD);
  cvt(Wc, wc, HD * EMBD);
  cvt(Wl0, wl0, HD * HD);
  cvt(Wr0, wr0, HD * HD);
  cvt(Wl1, wl1, HD * HD);
  cvt(Wr1, wr1, HD * HD);
  cvt(W1, w1h, HD * 2 * HD);
  if (x16) cvt(x_user, x16, N * TT * EMBD);

  hipMemsetAsync(cntS, 0, (size_t)N * 4, stream);
  hipMemsetAsync(cntD, 0, (size_t)N * 4, stream);
  edge_count<<<(int)((E + 255) / 256), 256, 0, stream>>>(srcI, cntS, E);
  edge_count<<<(int)((E + 255) / 256), 256, 0, stream>>>(dstI, cntD, E);

  if (x16)
    lstm_kernel<true><<<N / 16, 128, 0, stream>>>(x_user, x16, wih, whh, b_ih,
                                                  b_hh, u);
  else
    lstm_kernel<false><<<N / 16, 128, 0, stream>>>(x_user, nullptr, wih, whh,
                                                   b_ih, b_hh, u);
  linear128<<<N / 64, 128, 0, stream>>>(x_content, EMBD, wc, bc, c, 0);

  float *cu = u, *cc = c, *nu = u2, *nc = c2;
  const long long scatThreads = E * 32;
  const int scatBlocks = (int)((scatThreads + 255) / 256);
  for (int layer = 0; layer < 2; ++layer) {
    const _Float16* Wl = layer ? wl1 : wl0;
    const _Float16* Wr = layer ? wr1 : wr0;
    const float* bl = layer ? bl1 : bl0;
    // users -> content
    hipMemsetAsync(agg, 0, (size_t)N * HD * 4, stream);
    edge_scatter<<<scatBlocks, 256, 0, stream>>>(srcI, dstI, cu, agg, E);
    sage_gemm<<<N / 64, 128, 0, stream>>>(agg, cntD, cc, Wl, Wr, bl, nc);
    // content -> users
    hipMemsetAsync(agg, 0, (size_t)N * HD * 4, stream);
    edge_scatter<<<scatBlocks, 256, 0, stream>>>(dstI, srcI, cc, agg, E);
    sage_gemm<<<N / 64, 128, 0, stream>>>(agg, cntS, cu, Wl, Wr, bl, nu);
    float* t;
    t = cu; cu = nu; nu = t;
    t = cc; cc = nc; nc = t;
  }
  classifier<<<N / 64, 128, 0, stream>>>(cu, cc, w1h, b1, W2, b2,
                                         (float*)d_out);
}
